// MultiHeadAttentionLayer_16664473108907
// MI455X (gfx1250) — matrix operand order repaired and checked by reading, not compiled, not run on hardware
//
#include <hip/hip_runtime.h>

typedef __attribute__((ext_vector_type(16))) __bf16 v16bf;
typedef __attribute__((ext_vector_type(8)))  float  v8f;

union FragBf { v16bf v; uint4 q[2]; };

__device__ inline v8f v8f_zero() { v8f z; for (int i = 0; i < 8; ++i) z[i] = 0.f; return z; }

__device__ inline v8f wmma_bf16(const v16bf& a, const v16bf& b, const v8f& c) {
    return __builtin_amdgcn_wmma_f32_16x16x32_bf16(false, a, false, b, (short)0, c, false, false);
}

// Async L2->LDS copy (16B per lane), tracked by ASYNCcnt (CDNA5 path).
// saddr = uniform base, voffset = per-lane byte offset, lds_off = LDS byte addr.
__device__ inline void async_copy_b128(void* lds_dst, const void* base, unsigned byte_off) {
  unsigned lds_off = (unsigned)(uintptr_t)lds_dst;   // flat->LDS: addr[31:0]
  asm volatile("global_load_async_to_lds_b128 %0, %1, %2"
               :: "v"(lds_off), "v"(byte_off), "s"(base) : "memory");
}
__device__ inline void async_wait() {
  asm volatile("s_wait_asynccnt 0" ::: "memory");
}

#define D_MODEL 1024
#define N_HEADS 16
#define HEAD_DIM 64
#define SEQ 2048
#define NREL 65           // 2*MAX_REL+1
#define ROWS 4096         // B*T

// ---------------------------------------------------------------------------
// GEMM 1: C_bf16[M,N] = A_f32[M,K] @ W_f32[K,N] + bias  (Q,K,V projections)
// block 256 thr = 8 waves, tile 128x128, wave tile 32x64, K-step 32
// ---------------------------------------------------------------------------
__global__ __launch_bounds__(256) void gemm_f32_bf16out(
    const float* __restrict__ A, const float* __restrict__ W,
    const float* __restrict__ bias, __bf16* __restrict__ C,
    int M, int N, int K) {
  __shared__ __bf16 As[128][48];   // [m][k]
  __shared__ __bf16 Bs[128][48];   // [n][k]  (W tile transposed)
  const int tid  = threadIdx.x;
  const int lane = tid & 31, wid = tid >> 5;
  const int wm = wid >> 1, wn = wid & 1;
  const int half = lane >> 4, n16 = lane & 15;
  const int r0 = blockIdx.y * 128, n0 = blockIdx.x * 128;

  v8f acc[2][4];
  for (int ms = 0; ms < 2; ++ms)
    for (int ns = 0; ns < 4; ++ns) acc[ms][ns] = v8f_zero();

  for (int kk = 0; kk < K; kk += 32) {
    for (int i = 0; i < 4; ++i) {             // A tile 128x32 f32 -> bf16
      int lin = tid + i * 256;
      int row = lin >> 3, c4 = (lin & 7) * 4;
      float4 f = *(const float4*)(A + ((r0 + row) * K + kk + c4));
      As[row][c4 + 0] = (__bf16)f.x; As[row][c4 + 1] = (__bf16)f.y;
      As[row][c4 + 2] = (__bf16)f.z; As[row][c4 + 3] = (__bf16)f.w;
    }
    for (int i = 0; i < 4; ++i) {             // W tile 32x128 -> Bs[n][k]
      int lin = tid + i * 256;
      int k = lin >> 5, c4 = (lin & 31) * 4;
      float4 f = *(const float4*)(W + ((kk + k) * N + n0 + c4));
      Bs[c4 + 0][k] = (__bf16)f.x; Bs[c4 + 1][k] = (__bf16)f.y;
      Bs[c4 + 2][k] = (__bf16)f.z; Bs[c4 + 3][k] = (__bf16)f.w;
    }
    __syncthreads();
    FragBf a[2], b[4];
    for (int ms = 0; ms < 2; ++ms) {
      const __bf16* p = &As[wm * 32 + ms * 16 + n16][0];
      a[ms].q[0] = *(const uint4*)(p + half * 8);
      a[ms].q[1] = *(const uint4*)(p + 16 + half * 8);
    }
    for (int ns = 0; ns < 4; ++ns) {
      const __bf16* p = &Bs[wn * 64 + ns * 16 + n16][0];
      b[ns].q[0] = *(const uint4*)(p + half * 8);
      b[ns].q[1] = *(const uint4*)(p + 16 + half * 8);
    }
    for (int ms = 0; ms < 2; ++ms)
      for (int ns = 0; ns < 4; ++ns)
        acc[ms][ns] = wmma_bf16(a[ms].v, b[ns].v, acc[ms][ns]);
    __syncthreads();
  }
  const int colb = n0 + wn * 64 + n16;
  const int rowb = r0 + wm * 32 + 8 * half;
  for (int ms = 0; ms < 2; ++ms)
    for (int ns = 0; ns < 4; ++ns) {
      float bv = bias[colb + ns * 16];
      for (int i = 0; i < 8; ++i) {
        int idx = (rowb + ms * 16 + i) * N + colb + ns * 16;   // 32-bit index
        C[idx] = (__bf16)(acc[ms][ns][i] + bv);
      }
    }
}

// ---------------------------------------------------------------------------
// GEMM 2: C_f32[M,N] = A_bf16[M,K] @ W_f32[K,N] + bias  (output projection)
// A tile is a pure bf16 copy -> async L2->LDS loads (ASYNCcnt path).
// ---------------------------------------------------------------------------
__global__ __launch_bounds__(256) void gemm_bf16_f32out(
    const __bf16* __restrict__ A, const float* __restrict__ W,
    const float* __restrict__ bias, float* __restrict__ C,
    int M, int N, int K) {
  __shared__ __bf16 As[128][48];
  __shared__ __bf16 Bs[128][48];
  const int tid  = threadIdx.x;
  const int lane = tid & 31, wid = tid >> 5;
  const int wm = wid >> 1, wn = wid & 1;
  const int half = lane >> 4, n16 = lane & 15;
  const int r0 = blockIdx.y * 128, n0 = blockIdx.x * 128;

  v8f acc[2][4];
  for (int ms = 0; ms < 2; ++ms)
    for (int ns = 0; ns < 4; ++ns) acc[ms][ns] = v8f_zero();

  for (int kk = 0; kk < K; kk += 32) {
    for (int i = 0; i < 2; ++i) {             // A tile 128x32 bf16, async copy
      int lin = tid + i * 256;
      int row = lin >> 2, c8 = (lin & 3) * 8;
      async_copy_b128(&As[row][c8], A,
                      (unsigned)(((r0 + row) * K + kk + c8) * 2));
    }
    for (int i = 0; i < 4; ++i) {             // W tile needs f32->bf16 convert
      int lin = tid + i * 256;
      int k = lin >> 5, c4 = (lin & 31) * 4;
      float4 f = *(const float4*)(W + ((kk + k) * N + n0 + c4));
      Bs[c4 + 0][k] = (__bf16)f.x; Bs[c4 + 1][k] = (__bf16)f.y;
      Bs[c4 + 2][k] = (__bf16)f.z; Bs[c4 + 3][k] = (__bf16)f.w;
    }
    async_wait();
    __syncthreads();
    FragBf a[2], b[4];
    for (int ms = 0; ms < 2; ++ms) {
      const __bf16* p = &As[wm * 32 + ms * 16 + n16][0];
      a[ms].q[0] = *(const uint4*)(p + half * 8);
      a[ms].q[1] = *(const uint4*)(p + 16 + half * 8);
    }
    for (int ns = 0; ns < 4; ++ns) {
      const __bf16* p = &Bs[wn * 64 + ns * 16 + n16][0];
      b[ns].q[0] = *(const uint4*)(p + half * 8);
      b[ns].q[1] = *(const uint4*)(p + 16 + half * 8);
    }
    for (int ms = 0; ms < 2; ++ms)
      for (int ns = 0; ns < 4; ++ns)
        acc[ms][ns] = wmma_bf16(a[ms].v, b[ns].v, acc[ms][ns]);
    __syncthreads();
  }
  const int colb = n0 + wn * 64 + n16;
  const int rowb = r0 + wm * 32 + 8 * half;
  for (int ms = 0; ms < 2; ++ms)
    for (int ns = 0; ns < 4; ++ns) {
      float bv = bias[colb + ns * 16];
      for (int i = 0; i < 8; ++i) {
        int idx = (rowb + ms * 16 + i) * N + colb + ns * 16;
        C[idx] = acc[ms][ns][i] + bv;
      }
    }
}

// ---------------------------------------------------------------------------
// Rel-K projection: P[bh, q, r] = dot(Q[b,q,h,:], relK[r,:])   r in [0,65)
// grid (T/16, H, B), block 256; table + Q rows staged via async LDS loads
// ---------------------------------------------------------------------------
__global__ __launch_bounds__(256) void relk_proj(
    const __bf16* __restrict__ Q, const float* __restrict__ relk,
    float* __restrict__ P) {
  __shared__ float  rkL[NREL * HEAD_DIM];
  __shared__ __bf16 qL[16][HEAD_DIM];
  const int tid = threadIdx.x;
  const int h = blockIdx.y, b = blockIdx.z;
  const int bh = b * N_HEADS + h;
  const int qbase = blockIdx.x * 16;
  for (int c = tid; c < NREL * HEAD_DIM / 4; c += 256)      // 1040 x 16B chunks
    async_copy_b128(&rkL[c * 4], relk, (unsigned)(c * 16));
  for (int c = tid; c < 16 * 8; c += 256) {                 // 16 rows x 8 chunks
    int row = c >> 3, d8 = (c & 7) * 8;
    async_copy_b128(&qL[row][d8], Q,
                    (unsigned)(((b * SEQ + qbase + row) * D_MODEL + h * HEAD_DIM + d8) * 2));
  }
  async_wait();
  __syncthreads();
  for (int lin = tid; lin < 16 * NREL; lin += 256) {
    int row = lin / NREL, r = lin % NREL;
    float s = 0.f;
    for (int d = 0; d < HEAD_DIM; ++d) s += (float)qL[row][d] * rkL[r * HEAD_DIM + d];
    P[(bh * SEQ + qbase + row) * NREL + r] = s;
  }
}

// ---------------------------------------------------------------------------
// Fused flash attention with relative bias + rel-V bucket accumulation.
// grid (T/64, H, B), block 128 (4 waves); each wave owns 16 q-rows.
// ---------------------------------------------------------------------------
__global__ __launch_bounds__(128) void attn_flash(
    const __bf16* __restrict__ Q, const __bf16* __restrict__ Kt,
    const __bf16* __restrict__ V, const float* __restrict__ P,
    const float* __restrict__ relv, __bf16* __restrict__ Oh) {
  __shared__ __bf16 Vt[64][72];          // [d][k] transposed V tile
  __shared__ __bf16 pL[4][16][72];       // per-wave P staging for PV wmma
  __shared__ float  agg[64][NREL];       // rel-V bucket sums per q-row
  __shared__ float  rvL[NREL][HEAD_DIM];
  const int tid  = threadIdx.x;
  const int lane = tid & 31, wid = tid >> 5;
  const int half = lane >> 4, n16 = lane & 15;
  const int h = blockIdx.y, b = blockIdx.z;
  const int bh = b * N_HEADS + h;
  const int qbase = blockIdx.x * 64;

  for (int i = tid; i < 64 * NREL; i += 128) agg[i / NREL][i % NREL] = 0.f;
  for (int c = tid; c < NREL * HEAD_DIM / 4; c += 128)      // rel-V table, async
    async_copy_b128(&rvL[0][0] + c * 4, relv, (unsigned)(c * 16));

  // Q fragments (constant over k-loop): rows qbase + wid*16 + (lane%16)
  const int qrow = qbase + wid * 16 + n16;
  const __bf16* qp = Q + ((b * SEQ + qrow) * D_MODEL + h * HEAD_DIM);
  FragBf aq0, aq1;
  aq0.q[0] = *(const uint4*)(qp + half * 8);
  aq0.q[1] = *(const uint4*)(qp + 16 + half * 8);
  aq1.q[0] = *(const uint4*)(qp + 32 + half * 8);
  aq1.q[1] = *(const uint4*)(qp + 48 + half * 8);

  v8f o[4];
  for (int ns = 0; ns < 4; ++ns) o[ns] = v8f_zero();
  float l_r[8], m_r[8];
  for (int i = 0; i < 8; ++i) { l_r[i] = 0.f; m_r[i] = -1e30f; }
  const float* Pb = P + bh * SEQ * NREL;
  async_wait();
  __syncthreads();

  for (int kt = 0; kt < SEQ; kt += 64) {
    // stage V tile transposed: Vt[d][k] (needs reshape -> manual path)
    for (int i = 0; i < 4; ++i) {
      int lin = tid + i * 128;
      int k = lin >> 3, d8 = (lin & 7) * 8;
      uint4 u = *(const uint4*)(V + ((b * SEQ + kt + k) * D_MODEL + h * HEAD_DIM + d8));
      const __bf16* e = (const __bf16*)&u;
      for (int j = 0; j < 8; ++j) Vt[d8 + j][k] = e[j];
    }
    __syncthreads();

    float pv[4][8];
    int   ridx[4][8];
    for (int ns = 0; ns < 4; ++ns) {
      const __bf16* kp = Kt + ((b * SEQ + kt + ns * 16 + n16) * D_MODEL + h * HEAD_DIM);
      FragBf b0, b1;
      b0.q[0] = *(const uint4*)(kp + half * 8);
      b0.q[1] = *(const uint4*)(kp + 16 + half * 8);
      b1.q[0] = *(const uint4*)(kp + 32 + half * 8);
      b1.q[1] = *(const uint4*)(kp + 48 + half * 8);
      v8f s = v8f_zero();
      s = wmma_bf16(aq0.v, b0.v, s);
      s = wmma_bf16(aq1.v, b1.v, s);
      for (int i = 0; i < 8; ++i) {
        int qg = qbase + wid * 16 + i + 8 * half;
        int kg = kt + ns * 16 + n16;
        int dist = kg - qg;
        dist = dist < -32 ? -32 : (dist > 32 ? 32 : dist);
        int r = dist + 32;
        ridx[ns][i] = r;
        float bias = Pb[qg * NREL + r];
        pv[ns][i] = (s[i] + bias) * 0.125f;   // 1/sqrt(64)
      }
    }
    // online softmax per row (rows per-elem; cols across 16 lanes of a half)
    float alpha[8];
    for (int i = 0; i < 8; ++i) {
      float mx = pv[0][i];
      for (int ns = 1; ns < 4; ++ns) mx = fmaxf(mx, pv[ns][i]);
      for (int off = 1; off < 16; off <<= 1) mx = fmaxf(mx, __shfl_xor(mx, off, 32));
      float mnew = fmaxf(m_r[i], mx);
      float a = __expf(m_r[i] - mnew);
      float rs = 0.f;
      for (int ns = 0; ns < 4; ++ns) {
        float e = __expf(pv[ns][i] - mnew);
        pv[ns][i] = e; rs += e;
      }
      for (int off = 1; off < 16; off <<= 1) rs += __shfl_xor(rs, off, 32);
      l_r[i] = l_r[i] * a + rs;
      m_r[i] = mnew;
      alpha[i] = a;
    }
    for (int ns = 0; ns < 4; ++ns)
      for (int i = 0; i < 8; ++i) o[ns][i] *= alpha[i];
    // rel-V bucket accumulation: rescale then add (wave-private rows, DS in-order)
    if (n16 == 0) {
      for (int i = 0; i < 8; ++i) {
        int rowl = wid * 16 + i + 8 * half;
        float a = alpha[i];
        for (int r = 0; r < NREL; ++r) agg[rowl][r] *= a;
      }
    }
    for (int ns = 0; ns < 4; ++ns)
      for (int i = 0; i < 8; ++i)
        atomicAdd(&agg[wid * 16 + i + 8 * half][ridx[ns][i]], pv[ns][i]);
    // stage P as bf16 for PV wmma (C-layout -> A-layout via LDS)
    for (int ns = 0; ns < 4; ++ns)
      for (int i = 0; i < 8; ++i)
        pL[wid][i + 8 * half][ns * 16 + n16] = (__bf16)pv[ns][i];
    __syncthreads();

    FragBf ap0, ap1;
    const __bf16* pp = &pL[wid][n16][0];
    ap0.q[0] = *(const uint4*)(pp + half * 8);
    ap0.q[1] = *(const uint4*)(pp + 16 + half * 8);
    ap1.q[0] = *(const uint4*)(pp + 32 + half * 8);
    ap1.q[1] = *(const uint4*)(pp + 48 + half * 8);
    for (int ns = 0; ns < 4; ++ns) {
      const __bf16* vb = &Vt[ns * 16 + n16][0];
      FragBf bv0, bv1;
      bv0.q[0] = *(const uint4*)(vb + half * 8);
      bv0.q[1] = *(const uint4*)(vb + 16 + half * 8);
      bv1.q[0] = *(const uint4*)(vb + 32 + half * 8);
      bv1.q[1] = *(const uint4*)(vb + 48 + half * 8);
      o[ns] = wmma_bf16(ap0.v, bv0.v, o[ns]);
      o[ns] = wmma_bf16(ap1.v, bv1.v, o[ns]);
    }
    __syncthreads();
  }

  // epilogue: (o + agg @ relV) / l
  for (int ns = 0; ns < 4; ++ns)
    for (int i = 0; i < 8; ++i) {
      int rowl = wid * 16 + i + 8 * half;
      int qg = qbase + rowl;
      int d = ns * 16 + n16;
      float w2 = 0.f;
      for (int r = 0; r < NREL; ++r) w2 += agg[rowl][r] * rvL[r][d];
      float val = (o[ns][i] + w2) / l_r[i];
      Oh[(b * SEQ + qg) * D_MODEL + h * HEAD_DIM + d] = (__bf16)val;
    }
}

// ---------------------------------------------------------------------------
extern "C" void kernel_launch(void* const* d_in, const int* in_sizes, int n_in,
                              void* d_out, int out_size, void* d_ws, size_t ws_size,
                              hipStream_t stream) {
  (void)in_sizes; (void)n_in; (void)out_size; (void)ws_size;
  const float* query = (const float*)d_in[0];
  const float* key   = (const float*)d_in[1];
  const float* value = (const float*)d_in[2];
  const float* Wq = (const float*)d_in[3];  const float* bq = (const float*)d_in[4];
  const float* Wk = (const float*)d_in[5];  const float* bk = (const float*)d_in[6];
  const float* Wv = (const float*)d_in[7];  const float* bv = (const float*)d_in[8];
  const float* Wo = (const float*)d_in[9];  const float* bo = (const float*)d_in[10];
  const float* relk = (const float*)d_in[11];
  const float* relv = (const float*)d_in[12];
  float* out = (float*)d_out;

  // workspace layout
  __bf16* Qbf = (__bf16*)d_ws;
  __bf16* Kbf = Qbf + (size_t)ROWS * D_MODEL;
  __bf16* Vbf = Kbf + (size_t)ROWS * D_MODEL;
  float*  Prelk = (float*)(Vbf + (size_t)ROWS * D_MODEL);
  __bf16* Oheads = (__bf16*)(Prelk + (size_t)2 * N_HEADS * SEQ * NREL);

  dim3 gGemm(D_MODEL / 128, ROWS / 128);
  gemm_f32_bf16out<<<gGemm, 256, 0, stream>>>(query, Wq, bq, Qbf, ROWS, D_MODEL, D_MODEL);
  gemm_f32_bf16out<<<gGemm, 256, 0, stream>>>(key,   Wk, bk, Kbf, ROWS, D_MODEL, D_MODEL);
  gemm_f32_bf16out<<<gGemm, 256, 0, stream>>>(value, Wv, bv, Vbf, ROWS, D_MODEL, D_MODEL);

  relk_proj<<<dim3(SEQ / 16, N_HEADS, 2), 256, 0, stream>>>(Qbf, relk, Prelk);

  attn_flash<<<dim3(SEQ / 64, N_HEADS, 2), 128, 0, stream>>>(Qbf, Kbf, Vbf, Prelk, relv, Oheads);

  gemm_bf16_f32out<<<gGemm, 256, 0, stream>>>(Oheads, Wo, bo, out, ROWS, D_MODEL, D_MODEL);
}
